// TARGET_LSTM_46497315947053
// MI455X (gfx1250) — matrix operand order repaired and checked by reading, not compiled, not run on hardware
//
#include <hip/hip_runtime.h>
#include <hip/hip_bf16.h>

// ---------- problem constants ----------
#define VV 32000
#define EE 128
#define HH 128
#define BB 32
#define SS 32

typedef __attribute__((ext_vector_type(16))) __bf16 v16bf;
typedef __attribute__((ext_vector_type(8)))  __bf16 v8bf;
typedef __attribute__((ext_vector_type(8)))  float  v8f;

union V16U { v16bf v; v8bf h[2]; };

__device__ __forceinline__ float sigm_(float x) { return 1.0f / (1.0f + expf(-x)); }

// ---------- one-time prep: fused gate weights GT[512][256] bf16 + bg[512] ----------
__global__ void lstm_prep_gates(const float* Wi, const float* Ui, const float* bi,
                                const float* Wf, const float* Uf, const float* bf_,
                                const float* Wog, const float* Uog, const float* bog,
                                const float* Wc, const float* Uc, const float* bc,
                                __bf16* GT, float* bg) {
  int idx = blockIdx.x * 256 + threadIdx.x;           // 512*256 = 131072 exactly
  int n = idx >> 8;                                   // 0..511
  int k = idx & 255;                                  // 0..255
  int g = n >> 7;                                     // gate id
  int ncol = n & 127;
  const float* W[4] = {Wi, Wf, Wog, Wc};
  const float* U[4] = {Ui, Uf, Uog, Uc};
  float v = (k < 128) ? W[g][k * HH + ncol] : U[g][(k - 128) * HH + ncol];
  GT[n * 256 + k] = (__bf16)v;
  if (idx < 512) {
    const float* bsrc[4] = {bi, bf_, bog, bc};
    bg[idx] = bsrc[idx >> 7][idx & 127];
  }
}

// ---------- one-time prep: WoT[V][H] bf16 (transpose of Wo[H][V]) ----------
__global__ void lstm_prep_wot(const float* Wo, __bf16* WoT) {
  int idx = blockIdx.x * 256 + threadIdx.x;           // 16000*256 = 4,096,000 exactly
  int k = idx / VV;
  int n = idx - k * VV;                               // coalesced read of Wo
  WoT[(size_t)n * HH + k] = (__bf16)Wo[idx];
}

// ---------- one-time prep: zero h(bf16) and c(f32) ----------
__global__ void lstm_prep_zero(__bf16* hb, float* cst) {
  int idx = blockIdx.x * 256 + threadIdx.x;
  if (idx < 64 * HH) { hb[idx] = (__bf16)0.0f; cst[idx] = 0.0f; }
}

// ---------- per-step LSTM cell: 4 blocks, one 16-row M-tile each ----------
// global rows 0..31 = generation batch, rows 32..63 = teacher-forced batch
__global__ void lstm_cell(const int* x, const float* emb, const __bf16* GT, const float* bg,
                          __bf16* hb, float* cst, const int* nt, int t) {
  __shared__ __bf16 A[16 * 256];                      // [local row][k]: k<128 x_emb, k>=128 h_prev
  int tid = threadIdx.x;
  int m = blockIdx.x;                                 // M-tile 0..3
  for (int idx = tid; idx < 16 * 256; idx += 256) {
    int lr = idx >> 8, k = idx & 255;
    int grow = m * 16 + lr;                           // global row 0..63
    float v;
    if (k < 128) {
      int tok;
      if (grow < 32) tok = (t == 0) ? 0 : nt[grow];
      else           tok = (t == 0) ? 0 : x[(grow - 32) * SS + (t - 1)];
      v = emb[(size_t)tok * EE + k];
    } else {
      v = (float)hb[grow * HH + (k - 128)];
    }
    A[idx] = (__bf16)v;
  }
  __syncthreads();

  int wave = tid >> 5, lane = tid & 31;
  int l16 = lane & 15, hi = lane >> 4;
  int off = hi * 8, kb = hi * 16;
  int j = wave;                                       // N-subtile within H (0..7)

  v8f z = {0.f, 0.f, 0.f, 0.f, 0.f, 0.f, 0.f, 0.f};
  v8f acc[4] = {z, z, z, z};                          // i, f, o, c-gate tiles
  for (int kc = 0; kc < 8; ++kc) {                    // K = 256 in chunks of 32
    V16U ua;
    ua.h[0] = *(const v8bf*)&A[l16 * 256 + kc * 32 + off];
    ua.h[1] = *(const v8bf*)&A[l16 * 256 + kc * 32 + 16 + off];
#pragma unroll
    for (int g = 0; g < 4; ++g) {
      int n = g * 128 + j * 16 + l16;
      V16U ub;
      ub.h[0] = *(const v8bf*)&GT[n * 256 + kc * 32 + off];
      ub.h[1] = *(const v8bf*)&GT[n * 256 + kc * 32 + 16 + off];
      acc[g] = __builtin_amdgcn_wmma_f32_16x16x32_bf16(
          false, ua.v, false, ub.v, (short)0, acc[g], false, false);
    }
  }
  int col = j * 16 + l16;
#pragma unroll
  for (int r = 0; r < 8; ++r) {
    int row = m * 16 + r + hi * 8;
    float ig = sigm_(acc[0][r] + bg[col]);
    float fg = sigm_(acc[1][r] + bg[col + 128]);
    float og = sigm_(acc[2][r] + bg[col + 256]);
    float cg = tanhf(acc[3][r] + bg[col + 384]);
    float cold = cst[row * HH + col];
    float cnew = fg * cold + ig * cg;
    cst[row * HH + col] = cnew;
    hb[row * HH + col] = (__bf16)(og * tanhf(cnew));
  }
}

// ---------- per-step vocab projection: logits[64][V] = h @ Wo + bo ----------
// 500 blocks x 4 waves; each block covers 64 columns (4 N-tiles), each wave one M-tile.
// A fragments are loaded once per K-chunk and reused across the 4 N-tiles.
__global__ void lstm_vocab_gemm(const __bf16* hb, const __bf16* WoT, const float* bo,
                                float* logits) {
  int wave = threadIdx.x >> 5, lane = threadIdx.x & 31;
  int l16 = lane & 15, hi = lane >> 4;
  int off = hi * 8, kb = hi * 16;
  int n0 = blockIdx.x * 64;
  int arow = wave * 16 + l16;
  const __bf16* ap = hb + arow * HH;
  v8f z = {0.f, 0.f, 0.f, 0.f, 0.f, 0.f, 0.f, 0.f};
  v8f acc[4] = {z, z, z, z};
#pragma unroll
  for (int kc = 0; kc < 4; ++kc) {                    // K = 128 in chunks of 32
    V16U ua;
    ua.h[0] = *(const v8bf*)(ap + kc * 32 + off);
    ua.h[1] = *(const v8bf*)(ap + kc * 32 + 16 + off);
#pragma unroll
    for (int nb = 0; nb < 4; ++nb) {
      const __bf16* bp = WoT + (size_t)(n0 + nb * 16 + l16) * HH + kc * 32 + off;
      V16U ub;
      ub.h[0] = *(const v8bf*)bp;
      ub.h[1] = *(const v8bf*)(bp + 16);
      acc[nb] = __builtin_amdgcn_wmma_f32_16x16x32_bf16(
          false, ua.v, false, ub.v, (short)0, acc[nb], false, false);
    }
  }
#pragma unroll
  for (int nb = 0; nb < 4; ++nb) {
    int col = n0 + nb * 16 + l16;
    float bias = bo[col];
#pragma unroll
    for (int r = 0; r < 8; ++r) {
      int row = wave * 16 + r + hi * 8;
      logits[(size_t)row * VV + col] = acc[nb][r] + bias;
    }
  }
}

// ---------- per-step softmax/argmax: one block (512 threads) per row ----------
__global__ void lstm_softmax(const float* logits, const int* x, float* out,
                             int* nt, float* logp, int t) {
  int row = blockIdx.x, tid = threadIdx.x;
  const float* L = logits + (size_t)row * VV;
  __shared__ float sv[512];
  __shared__ int   si[512];
  // pass 1: max + first-occurrence argmax
  float m = -3.4e38f; int mi = 0;
  for (int v = tid; v < VV; v += 512) { float l = L[v]; if (l > m) { m = l; mi = v; } }
  sv[tid] = m; si[tid] = mi; __syncthreads();
  for (int s = 256; s > 0; s >>= 1) {
    if (tid < s) {
      float v2 = sv[tid + s]; int i2 = si[tid + s];
      if (v2 > sv[tid] || (v2 == sv[tid] && i2 < si[tid])) { sv[tid] = v2; si[tid] = i2; }
    }
    __syncthreads();
  }
  float rmax = sv[0]; int rarg = si[0]; __syncthreads();
  // pass 2: sum of exp
  float s0 = 0.f;
  for (int v = tid; v < VV; v += 512) s0 += expf(L[v] - rmax);
  sv[tid] = s0; __syncthreads();
  for (int s = 256; s > 0; s >>= 1) { if (tid < s) sv[tid] += sv[tid + s]; __syncthreads(); }
  float rsum = sv[0];
  // pass 3: outputs
  if (row < 32) {                                     // generation rows
    if (tid == 0) {
      nt[row] = rarg;
      out[row * SS + t] = (float)rarg;                // gen_x
      out[BB * SS + row * SS + t] = 1.0f / rsum;      // gen_o (exp(max-max)/sum)
    }
  } else {                                            // teacher-forced rows
    int b = row - 32;
    float inv = 1.0f / rsum;
    float* P = out + 2 * BB * SS + ((size_t)(b * SS + t)) * VV;
    for (int v = tid; v < VV; v += 512)
      __builtin_nontemporal_store(expf(L[v] - rmax) * inv, P + v);
    if (tid == 0) {
      int tok = x[b * SS + t];
      logp[b * SS + t] = (L[tok] - rmax) - logf(rsum);
    }
  }
}

// ---------- final loss (deterministic tree reduction) ----------
__global__ void lstm_loss(const float* logp, float* out) {
  __shared__ float s[256];
  int tid = threadIdx.x;
  float a = 0.f;
  for (int i = tid; i < BB * SS; i += 256) a += logp[i];
  s[tid] = a; __syncthreads();
  for (int k = 128; k > 0; k >>= 1) { if (tid < k) s[tid] += s[tid + k]; __syncthreads(); }
  if (tid == 0) out[2 * BB * SS + (size_t)BB * SS * VV] = -s[0] / (float)(BB * SS);
}

// ---------- launch ----------
extern "C" void kernel_launch(void* const* d_in, const int* in_sizes, int n_in,
                              void* d_out, int out_size, void* d_ws, size_t ws_size,
                              hipStream_t stream) {
  const int*   x   = (const int*)  d_in[0];
  const float* emb = (const float*)d_in[1];
  const float* Wi  = (const float*)d_in[2];
  const float* Ui  = (const float*)d_in[3];
  const float* bi  = (const float*)d_in[4];
  const float* Wf  = (const float*)d_in[5];
  const float* Uf  = (const float*)d_in[6];
  const float* bf_ = (const float*)d_in[7];
  const float* Wog = (const float*)d_in[8];
  const float* Uog = (const float*)d_in[9];
  const float* bog = (const float*)d_in[10];
  const float* Wc  = (const float*)d_in[11];
  const float* Uc  = (const float*)d_in[12];
  const float* bc  = (const float*)d_in[13];
  const float* Wo  = (const float*)d_in[14];
  const float* bo  = (const float*)d_in[15];
  (void)in_sizes; (void)n_in; (void)out_size; (void)ws_size;

  char* ws = (char*)d_ws;
  __bf16* WoT    = (__bf16*)(ws + 0);                  //  8,192,000 B
  float*  logits = (float*) (ws + 8192000);            //  8,192,000 B
  __bf16* GT     = (__bf16*)(ws + 16384000);           //    262,144 B
  float*  bg     = (float*) (ws + 16646144);           //      2,048 B
  __bf16* hb     = (__bf16*)(ws + 16648192);           //     16,384 B
  float*  cst    = (float*) (ws + 16664576);           //     32,768 B
  int*    nt     = (int*)   (ws + 16697344);           //        128 B
  float*  logp   = (float*) (ws + 16697472);           //      4,096 B
  float*  out    = (float*)d_out;

  lstm_prep_gates<<<512, 256, 0, stream>>>(Wi, Ui, bi, Wf, Uf, bf_, Wog, Uog, bog,
                                           Wc, Uc, bc, GT, bg);
  lstm_prep_wot<<<16000, 256, 0, stream>>>(Wo, WoT);
  lstm_prep_zero<<<32, 256, 0, stream>>>(hb, cst);

  for (int t = 0; t < SS; ++t) {
    lstm_cell<<<4, 256, 0, stream>>>(x, emb, GT, bg, hb, cst, nt, t);
    lstm_vocab_gemm<<<VV / 64, 128, 0, stream>>>(hb, WoT, bo, logits);
    lstm_softmax<<<64, 512, 0, stream>>>(logits, x, out, nt, logp, t);
  }
  lstm_loss<<<1, 256, 0, stream>>>(logp, out);
}
